// LiquidNet_8254927143576
// MI455X (gfx1250) — matrix operand order repaired and checked by reading, not compiled, not run on hardware
//
#include <hip/hip_runtime.h>
#include <hip/hip_bf16.h>

typedef __attribute__((ext_vector_type(16))) _Float16 v16h;
typedef __attribute__((ext_vector_type(8)))  _Float16 v8h;
typedef __attribute__((ext_vector_type(8)))  float    v8f;

#define B_ 128
#define S_ 2048
#define I_ 128
#define H_ 256
#define O_ 128

// workspace layout in halves: transposed f16 weights, WT[n][k] contiguous in k
#define WT_L0_SZ   (H_ * 384)                    // per gate, layer0 (K=384)
#define WT_L1_OFF  (4 * WT_L0_SZ)                // 393216
#define WT_L1_SZ   (H_ * 512)                    // per gate, layer1 (K=512)
#define WT_OUT_OFF (WT_L1_OFF + 4 * WT_L1_SZ)    // 917504
#define WT_TOTAL   (WT_OUT_OFF + O_ * H_)        // 950272 halves (~1.9MB)

// ---------------------------------------------------------------------------
// Branch-free transcendentals on the hardware TRANS pipe (v_exp_f32/v_rcp_f32).
// These co-issue with WMMA (XDL ops are tracked as TRANS) and avoid the OCML
// slow-path exec-mask branching that serialized the activation phase.
// ---------------------------------------------------------------------------
__device__ __forceinline__ float fast_tanh(float x) {
    // tanh(x) = 1 - 2/(exp(2x)+1); exp2 saturation gives correct +-1 limits.
    float e = __builtin_amdgcn_exp2f(x * 2.88539008177f);   // exp(2x)
    return 1.0f - 2.0f * __builtin_amdgcn_rcpf(e + 1.0f);
}
__device__ __forceinline__ float fast_sigmoid(float x) {
    float e = __builtin_amdgcn_exp2f(-x * 1.44269504089f);  // exp(-x)
    return __builtin_amdgcn_rcpf(1.0f + e);
}

// ---------------------------------------------------------------------------
// Pre-pass: W[K][N] f32 -> WT[N][K] f16 (so B-fragments are contiguous per row)
// ---------------------------------------------------------------------------
__global__ void transpose_f32_to_f16(const float* __restrict__ W,
                                     _Float16* __restrict__ WT, int K, int N) {
    int i = blockIdx.x * blockDim.x + threadIdx.x;
    int total = K * N;
    if (i < total) {
        int k = i / N, n = i - k * N;
        WT[(size_t)n * K + k] = (_Float16)W[i];
    }
}

// A fragment: 16x32 f16, lane<16: M=lane, K = k0+0..15
//             lane>=16: M=lane-16, K = k0+16..31
__device__ __forceinline__ v16h load_a_frag(const _Float16* zrow, int k0, int hi16) {
    int base = k0 + (hi16 << 4);
    union { v16h v; v8h h[2]; } u;
    u.h[0] = *(const v8h*)(zrow + base);
    u.h[1] = *(const v8h*)(zrow + base + 8);
    return u.v;
}

// One CfC layer tile: 4 gate GEMMs ([16,KD]x[KD,16]) sharing the A fragment,
// then the closed-form cell update. Output: h for rows m=r+8*hi16, col n.
template <int KD>
__device__ __forceinline__ void cfc_cell_tile(
    const _Float16* zc,         // LDS z tile [16][KD] f16
    const _Float16* wg,         // global: gate g at wg + g*(256*KD), row n at +n*KD
    const _Float16* gbl,        // LDS biases: gbl + g*256 + n
    const float* dts,           // LDS per-row dt
    int ntile, int lane, float hout[8]) {
    const int l15 = lane & 15;
    const int hi16 = lane >> 4;
    const int n = ntile * 16 + l15;

    v8f a0, a1, a2, a3;
    {
        float b0 = (float)gbl[0 * 256 + n];
        float b1 = (float)gbl[1 * 256 + n];
        float b2 = (float)gbl[2 * 256 + n];
        float b3 = (float)gbl[3 * 256 + n];
#pragma unroll
        for (int r = 0; r < 8; ++r) { a0[r] = b0; a1[r] = b1; a2[r] = b2; a3[r] = b3; }
    }

    const _Float16* zrow = zc + l15 * KD;
    const _Float16* w0 = wg + (size_t)0 * (256 * KD) + (size_t)n * KD;
    const _Float16* w1 = wg + (size_t)1 * (256 * KD) + (size_t)n * KD;
    const _Float16* w2 = wg + (size_t)2 * (256 * KD) + (size_t)n * KD;
    const _Float16* w3 = wg + (size_t)3 * (256 * KD) + (size_t)n * KD;

#pragma unroll 2
    for (int k0 = 0; k0 < KD; k0 += 32) {
        v16h a = load_a_frag(zrow, k0, hi16);
        int ko = k0 + (hi16 << 4);   // B: lanes<16 K=k0..k0+15, lanes>=16 K=k0+16..k0+31
        v16h b0 = *(const v16h*)(w0 + ko);
        v16h b1 = *(const v16h*)(w1 + ko);
        v16h b2 = *(const v16h*)(w2 + ko);
        v16h b3 = *(const v16h*)(w3 + ko);
        a0 = __builtin_amdgcn_wmma_f32_16x16x32_f16(false, a, false, b0, (short)0, a0, false, false);
        a1 = __builtin_amdgcn_wmma_f32_16x16x32_f16(false, a, false, b1, (short)0, a1, false, false);
        a2 = __builtin_amdgcn_wmma_f32_16x16x32_f16(false, a, false, b2, (short)0, a2, false, false);
        a3 = __builtin_amdgcn_wmma_f32_16x16x32_f16(false, a, false, b3, (short)0, a3, false, false);
    }

#pragma unroll
    for (int r = 0; r < 8; ++r) {
        int m = r + (hi16 << 3);
        float dtv = dts[m];
        float ff1 = fast_tanh(a0[r]);
        float ff2 = fast_tanh(a1[r]);
        float ti  = fast_sigmoid(a2[r] * dtv + a3[r]);
        hout[r] = ff1 * (1.0f - ti) + ti * ff2;
    }
}

// ---------------------------------------------------------------------------
// Persistent CfC kernel: 1 block = 16 batch rows, full time loop, 16 waves.
// ---------------------------------------------------------------------------
__launch_bounds__(512)
__global__ void cfc_persistent_kernel(
    const float* __restrict__ x, const float* __restrict__ dt,
    const _Float16* __restrict__ wt,
    const float* __restrict__ bf1_0, const float* __restrict__ bf2_0,
    const float* __restrict__ bta_0, const float* __restrict__ btb_0,
    const float* __restrict__ bf1_1, const float* __restrict__ bf2_1,
    const float* __restrict__ bta_1, const float* __restrict__ btb_1,
    const float* __restrict__ b_out, float* __restrict__ out) {
    __shared__ _Float16 z0[2][16 * 384];   // [x(128) | h0(256)] double-buffered
    __shared__ _Float16 z1[2][16 * 512];   // [h0(256) | h1(256)] double-buffered
    __shared__ float    dts[16];
    __shared__ _Float16 gb[8][H_];         // gate biases, [layer*4+gate][n]
    __shared__ _Float16 ob[O_];

    const int tid  = threadIdx.x;
    const int lane = tid & 31;
    const int wid  = tid >> 5;             // 16 waves: one H-tile of 16 each
    const int l15  = lane & 15;
    const int hi16 = lane >> 4;
    const int b0   = blockIdx.x * 16;

    // preload biases to LDS
    for (int i = tid; i < H_; i += 512) {
        gb[0][i] = (_Float16)bf1_0[i]; gb[1][i] = (_Float16)bf2_0[i];
        gb[2][i] = (_Float16)bta_0[i]; gb[3][i] = (_Float16)btb_0[i];
        gb[4][i] = (_Float16)bf1_1[i]; gb[5][i] = (_Float16)bf2_1[i];
        gb[6][i] = (_Float16)bta_1[i]; gb[7][i] = (_Float16)btb_1[i];
    }
    for (int i = tid; i < O_; i += 512) ob[i] = (_Float16)b_out[i];

    // h0(-1) = h1(-1) = 0
    for (int i = tid; i < 16 * 256; i += 512) {
        int m = i >> 8, n = i & 255;
        z0[0][m * 384 + 128 + n] = (_Float16)0.0f;
        z1[0][m * 512 + 256 + n] = (_Float16)0.0f;
    }
    __syncthreads();

    for (int t = 0; t < S_; ++t) {
        const int cur = t & 1;
        const int nxt = cur ^ 1;

        // stage x_t (f32 -> f16) and dt_t into LDS
        for (int i = tid; i < 16 * I_; i += 512) {
            int m = i >> 7, c = i & 127;
            z0[cur][m * 384 + c] =
                (_Float16)x[((size_t)(b0 + m) * S_ + (size_t)t) * I_ + c];
        }
        if (tid < 16) dts[tid] = dt[(size_t)(b0 + tid) * S_ + (size_t)t];
        __syncthreads();

        // ---- layer 0: z0[cur] (K=384) -> h0 ----
        float h0v[8];
        cfc_cell_tile<384>(z0[cur], wt, &gb[0][0], dts, wid, lane, h0v);
        {
            int n = wid * 16 + l15;
#pragma unroll
            for (int r = 0; r < 8; ++r) {
                int m = r + (hi16 << 3);
                _Float16 hv = (_Float16)h0v[r];
                z1[cur][m * 512 + n]       = hv;   // layer-1 input this step
                z0[nxt][m * 384 + 128 + n] = hv;   // layer-0 recurrence next step
            }
        }
        __syncthreads();

        // ---- layer 1: z1[cur] (K=512) -> h1 ----
        float h1v[8];
        cfc_cell_tile<512>(z1[cur], wt + WT_L1_OFF, &gb[4][0], dts, wid, lane, h1v);
        {
            int n = wid * 16 + l15;
#pragma unroll
            for (int r = 0; r < 8; ++r) {
                int m = r + (hi16 << 3);
                z1[nxt][m * 512 + 256 + n] = (_Float16)h1v[r];  // h1 for next step
            }
        }
        __syncthreads();

        // ---- output: h1 @ W_out + b_out (waves 0..7, O=128) ----
        if (wid < 8) {
            int n = wid * 16 + l15;
            v8f acc;
            {
                float bv = (float)ob[n];
#pragma unroll
                for (int r = 0; r < 8; ++r) acc[r] = bv;
            }
            const _Float16* zrow = &z1[nxt][l15 * 512 + 256];  // h1(t)
            const _Float16* wr   = wt + WT_OUT_OFF + (size_t)n * 256;
#pragma unroll 2
            for (int k0 = 0; k0 < 256; k0 += 32) {
                v16h a = load_a_frag(zrow, k0, hi16);
                v16h b = *(const v16h*)(wr + k0 + (hi16 << 4));
                acc = __builtin_amdgcn_wmma_f32_16x16x32_f16(false, a, false, b,
                                                             (short)0, acc, false, false);
            }
#pragma unroll
            for (int r = 0; r < 8; ++r) {
                int m = r + (hi16 << 3);
                out[((size_t)(b0 + m) * S_ + (size_t)t) * O_ + n] = acc[r];
            }
        }
        __syncthreads();
    }
}

// ---------------------------------------------------------------------------
extern "C" void kernel_launch(void* const* d_in, const int* in_sizes, int n_in,
                              void* d_out, int out_size, void* d_ws, size_t ws_size,
                              hipStream_t stream) {
    const float* x  = (const float*)d_in[0];
    const float* dt = (const float*)d_in[1];
    _Float16* wt = (_Float16*)d_ws;

    // convert + transpose all weights into f16 workspace
    struct Mat { int idx; size_t off; int K, N; };
    const Mat mats[9] = {
        {2,  (size_t)0 * WT_L0_SZ,            384, 256},  // W_ff1_0
        {4,  (size_t)1 * WT_L0_SZ,            384, 256},  // W_ff2_0
        {6,  (size_t)2 * WT_L0_SZ,            384, 256},  // W_ta_0
        {8,  (size_t)3 * WT_L0_SZ,            384, 256},  // W_tb_0
        {10, WT_L1_OFF + (size_t)0 * WT_L1_SZ, 512, 256}, // W_ff1_1
        {12, WT_L1_OFF + (size_t)1 * WT_L1_SZ, 512, 256}, // W_ff2_1
        {14, WT_L1_OFF + (size_t)2 * WT_L1_SZ, 512, 256}, // W_ta_1
        {16, WT_L1_OFF + (size_t)3 * WT_L1_SZ, 512, 256}, // W_tb_1
        {18, (size_t)WT_OUT_OFF,              256, 128},  // W_out
    };
    for (int i = 0; i < 9; ++i) {
        int total = mats[i].K * mats[i].N;
        transpose_f32_to_f16<<<(total + 255) / 256, 256, 0, stream>>>(
            (const float*)d_in[mats[i].idx], wt + mats[i].off, mats[i].K, mats[i].N);
    }

    cfc_persistent_kernel<<<dim3(B_ / 16), dim3(512), 0, stream>>>(
        x, dt, wt,
        (const float*)d_in[3],  (const float*)d_in[5],
        (const float*)d_in[7],  (const float*)d_in[9],
        (const float*)d_in[11], (const float*)d_in[13],
        (const float*)d_in[15], (const float*)d_in[17],
        (const float*)d_in[19], (float*)d_out);
}
